// Attention_65996467470934
// MI455X (gfx1250) — matrix operand order repaired and checked by reading, not compiled, not run on hardware
//
#include <hip/hip_runtime.h>
#include <hip/hip_bf16.h>

// ---------------------------------------------------------------------------
// CDNA5 (gfx1250) transformer block: preLN attention (qk-norm, f32 logits) + MLP
// All GEMMs via v_wmma_f32_16x16x32_f16 (wave32). fp32 accumulation everywhere.
// GEMM: 128x128 block tile, 8 waves, wave = 32x64 (2x4 WMMA), K-step 32.
// ---------------------------------------------------------------------------

typedef __attribute__((ext_vector_type(16))) _Float16 v16h;
typedef __attribute__((ext_vector_type(8)))  float    v8f;

struct __align__(16) h8 { _Float16 h[8]; };

__device__ __forceinline__ v8f wmma16(v16h a, v16h b, v8f c) {
  // 8 args: (neg_a, A, neg_b, B, c_mod, C, reuse_a, reuse_b)
  return __builtin_amdgcn_wmma_f32_16x16x32_f16(false, a, false, b, (short)0, c,
                                                false, false);
}

union FragU { v16h v; uint4 u[2]; };

// A fragment (16x32 f16, row-major tile in LDS, leading dim `ld` halfs):
// lane holds row M = l16; K indices kofs+hid*8..+7 (VGPR0-3) and
// kofs+16+hid*8..+7 (VGPR4-7)  [ISA 7.12.2, 16-bit A 16x32 table]
__device__ __forceinline__ v16h load_frag_a(const _Float16* lds, int row, int ld,
                                            int kofs, int hid) {
  FragU f;
  const _Float16* p = lds + row * ld + kofs;
  f.u[0] = *(const uint4*)(p + hid * 8);
  f.u[1] = *(const uint4*)(p + 16 + hid * 8);
  return f.v;
}

// B fragment (32x16 f16) from *transposed* LDS store Bt[n][k] (leading dim `ld`):
// lane holds column n = l16; K indices kofs+hid*8..+7 (VGPR0-3) and
// kofs+16+hid*8..+7 (VGPR4-7), same order as the A fragment
__device__ __forceinline__ v16h load_frag_b(const _Float16* lds, int row, int ld,
                                            int kofs, int hid) {
  FragU f;
  const _Float16* p = lds + row * ld + kofs;
  f.u[0] = *(const uint4*)(p + hid * 8);
  f.u[1] = *(const uint4*)(p + 16 + hid * 8);
  return f.v;
}

__device__ __forceinline__ float gelu_tanh(float x) {
  const float c = 0.7978845608028654f;  // sqrt(2/pi)
  return 0.5f * x * (1.0f + tanhf(c * (x + 0.044715f * x * x * x)));
}

// --------------------------- elementwise f32 -> f16 -------------------------
__global__ void f2h_kernel(const float* __restrict__ s, _Float16* __restrict__ d,
                           int n) {
  int i = blockIdx.x * blockDim.x + threadIdx.x;
  if (i < n) d[i] = (_Float16)s[i];
}

// ------------------------- row RMS-norm (f32 -> f16) ------------------------
// y = x * rsqrt(mean(x^2) + eps) * scale ; one block per row, C elems.
__global__ __launch_bounds__(256) void rmsnorm_rows(const float* __restrict__ x,
                                                    const float* __restrict__ scale,
                                                    _Float16* __restrict__ out,
                                                    int C) {
  __shared__ float red[8];
  const size_t row = blockIdx.x;
  const float* xr = x + row * (size_t)C;
  float ss = 0.f;
  for (int i = threadIdx.x; i < C; i += 256) { float v = xr[i]; ss += v * v; }
  #pragma unroll
  for (int m = 16; m; m >>= 1) ss += __shfl_xor(ss, m, 32);
  if ((threadIdx.x & 31) == 0) red[threadIdx.x >> 5] = ss;
  __syncthreads();
  float tot = 0.f;
  #pragma unroll
  for (int i = 0; i < 8; ++i) tot += red[i];
  const float r = rsqrtf(tot / (float)C + 1e-6f);
  for (int i = threadIdx.x; i < C; i += 256)
    out[row * (size_t)C + i] = (_Float16)(xr[i] * r * scale[i]);
}

// ---------------- per-head RMS-norm on q/k (f16 in place, D=64) -------------
// block = one (b,s) row; thread t: head = t/16, lane16 = t%16 owns 4 d's.
__global__ __launch_bounds__(256) void qk_norm_kernel(_Float16* __restrict__ q,
                                                      const float* __restrict__ scale,
                                                      float extra) {
  const size_t row = blockIdx.x;            // b*S + s
  const int h = threadIdx.x >> 4;
  const int j = threadIdx.x & 15;
  _Float16* p = q + ((row * 16 + h) * 64) + j * 4;
  float v0 = (float)p[0], v1 = (float)p[1], v2 = (float)p[2], v3 = (float)p[3];
  float ss = v0 * v0 + v1 * v1 + v2 * v2 + v3 * v3;
  #pragma unroll
  for (int m = 8; m; m >>= 1) ss += __shfl_xor(ss, m, 32);  // within 16-group
  const float r = rsqrtf(ss * (1.0f / 64.0f) + 1e-6f) * extra;
  p[0] = (_Float16)(v0 * r * scale[j * 4 + 0]);
  p[1] = (_Float16)(v1 * r * scale[j * 4 + 1]);
  p[2] = (_Float16)(v2 * r * scale[j * 4 + 2]);
  p[3] = (_Float16)(v3 * r * scale[j * 4 + 3]);
}

// --------------------------- generic WMMA GEMM ------------------------------
// C[M,N] = A[M,K] @ B[K,N]; A,B f16 row-major. 256 threads = 8 waves,
// 128x128 block tile; wave = 32x64 subtile (2x4 WMMA accums), K-step 32.
// EPI: 0 -> Ch = (f16)acc ; 1 -> Cf = acc + Res ; 2 -> Ch = (f16)gelu(acc)
template <int EPI>
__global__ __launch_bounds__(256) void gemm_kernel(const _Float16* __restrict__ A,
                                                   const _Float16* __restrict__ Bm,
                                                   const float* __restrict__ Res,
                                                   float* __restrict__ Cf,
                                                   _Float16* __restrict__ Ch,
                                                   int M, int N, int K) {
  __shared__ _Float16 As[128 * 32];   // A tile, row-major [m][k]   (8 KB)
  __shared__ _Float16 Bts[128 * 32];  // B tile transposed [n][k]   (8 KB)
  const int m0 = blockIdx.y * 128, n0 = blockIdx.x * 128;
  const int tid = threadIdx.x, wave = tid >> 5, lane = tid & 31;
  const int l16 = lane & 15, hid = (lane >> 4) & 1;
  const int wm = (wave >> 1) * 32;   // 0,32,64,96
  const int wn = (wave & 1) * 64;    // 0,64
  v8f acc[2][4] = {};

  for (int kk = 0; kk < K; kk += 32) {
    #pragma unroll
    for (int c = tid; c < 512; c += 256) {      // 128 rows x 32 halfs
      int row = c >> 2, cb = (c & 3) * 8;
      const _Float16* gsrc = &A[(size_t)(m0 + row) * K + kk + cb];
      *(uint4*)&As[row * 32 + cb] = *(const uint4*)gsrc;
      if (kk + 32 < K) __builtin_prefetch(gsrc + 32, 0, 0);  // global_prefetch_b8
    }
    #pragma unroll
    for (int c = tid; c < 512; c += 256) {      // 32 k-rows x 128 n, transpose
      int k = c >> 4, nb = (c & 15) * 8;
      h8 v = *(const h8*)&Bm[(size_t)(kk + k) * N + n0 + nb];
      #pragma unroll
      for (int e = 0; e < 8; ++e) Bts[(nb + e) * 32 + k] = v.h[e];
    }
    __syncthreads();
    v16h a0 = load_frag_a(As, wm + l16, 32, 0, hid);
    v16h a1 = load_frag_a(As, wm + 16 + l16, 32, 0, hid);
    #pragma unroll
    for (int ni = 0; ni < 4; ++ni) {
      v16h b = load_frag_b(Bts, wn + ni * 16 + l16, 32, 0, hid);
      acc[0][ni] = wmma16(a0, b, acc[0][ni]);
      acc[1][ni] = wmma16(a1, b, acc[1][ni]);
    }
    __syncthreads();
  }

  #pragma unroll
  for (int mi = 0; mi < 2; ++mi) {
    #pragma unroll
    for (int ni = 0; ni < 4; ++ni) {
      #pragma unroll
      for (int r = 0; r < 8; ++r) {
        const int row = m0 + wm + mi * 16 + (hid ? 8 + r : r);
        const int col = n0 + wn + ni * 16 + l16;
        const float v = acc[mi][ni][r];
        if (EPI == 0) {
          Ch[(size_t)row * N + col] = (_Float16)v;
        } else if (EPI == 1) {
          Cf[(size_t)row * N + col] = v + Res[(size_t)row * N + col];
        } else {
          Ch[(size_t)row * N + col] = (_Float16)gelu_tanh(v);
        }
      }
    }
  }
}

// ----------------------------- flash attention ------------------------------
// grid (S/64, H, B); 128 threads = 4 waves; wave owns 16 q rows x D=64 output.
// Online softmax over 64-key steps; logits and accum in f32 via WMMA f16.
__global__ __launch_bounds__(128) void attn_kernel(const _Float16* __restrict__ Q,
                                                   const _Float16* __restrict__ K,
                                                   const _Float16* __restrict__ V,
                                                   _Float16* __restrict__ O,
                                                   int S, int H) {
  const int D = 64;
  __shared__ _Float16 qs[64 * 64];   // [qrow][d]
  __shared__ _Float16 ks[64 * 64];   // [krow][d]  == transposed-B layout for q.k^T
  __shared__ _Float16 vt[64 * 64];   // [d][krow]  == transposed-B layout for P.V
  __shared__ _Float16 ps[4 * 16 * 64];  // per-wave P tile [16][64]
  const int q0 = blockIdx.x * 64, h = blockIdx.y, b = blockIdx.z;
  const int tid = threadIdx.x, wave = tid >> 5, lane = tid & 31;
  const int l16 = lane & 15, hid = (lane >> 4) & 1;
  const size_t base = (((size_t)b * S) * H + h) * D;  // element ((b*S+s)*H+h)*D+d
  const size_t rstr = (size_t)H * D;

  for (int c = tid; c < 64 * 8; c += 128) {
    int row = c >> 3, db = (c & 7) * 8;
    *(uint4*)&qs[row * 64 + db] = *(const uint4*)&Q[base + (size_t)(q0 + row) * rstr + db];
  }
  v8f o0 = {}, o1 = {}, o2 = {}, o3 = {};
  float mrow[8], lrow[8];
  #pragma unroll
  for (int r = 0; r < 8; ++r) { mrow[r] = -1e30f; lrow[r] = 0.f; }
  __syncthreads();

  for (int j0 = 0; j0 < S; j0 += 64) {
    for (int c = tid; c < 64 * 8; c += 128) {
      int row = c >> 3, db = (c & 7) * 8;
      *(uint4*)&ks[row * 64 + db] = *(const uint4*)&K[base + (size_t)(j0 + row) * rstr + db];
    }
    for (int c = tid; c < 64 * 8; c += 128) {
      int j = c >> 3, db = (c & 7) * 8;
      h8 vv = *(const h8*)&V[base + (size_t)(j0 + j) * rstr + db];
      #pragma unroll
      for (int e = 0; e < 8; ++e) vt[(db + e) * 64 + j] = vv.h[e];
    }
    __syncthreads();

    // scores: 16(q) x 64(k), K-contraction over d (two 32-chunks)
    v8f s0 = {}, s1 = {}, s2 = {}, s3 = {};
    #pragma unroll
    for (int kc = 0; kc < 64; kc += 32) {
      v16h aq = load_frag_a(qs, wave * 16 + l16, 64, kc, hid);
      v16h kb0 = load_frag_b(ks, 0 + l16, 64, kc, hid);
      v16h kb1 = load_frag_b(ks, 16 + l16, 64, kc, hid);
      v16h kb2 = load_frag_b(ks, 32 + l16, 64, kc, hid);
      v16h kb3 = load_frag_b(ks, 48 + l16, 64, kc, hid);
      s0 = wmma16(aq, kb0, s0);
      s1 = wmma16(aq, kb1, s1);
      s2 = wmma16(aq, kb2, s2);
      s3 = wmma16(aq, kb3, s3);
    }

    // online softmax; acc layout: vgpr r -> row (hid?8+r:r), col = tile*16+l16
    #pragma unroll
    for (int r = 0; r < 8; ++r) {
      float mx = fmaxf(fmaxf(s0[r], s1[r]), fmaxf(s2[r], s3[r]));
      #pragma unroll
      for (int m = 8; m; m >>= 1) mx = fmaxf(mx, __shfl_xor(mx, m, 32));
      const float mnew = fmaxf(mrow[r], mx);
      const float alpha = __expf(mrow[r] - mnew);
      const float p0 = __expf(s0[r] - mnew);
      const float p1 = __expf(s1[r] - mnew);
      const float p2 = __expf(s2[r] - mnew);
      const float p3 = __expf(s3[r] - mnew);
      float rs = p0 + p1 + p2 + p3;
      #pragma unroll
      for (int m = 8; m; m >>= 1) rs += __shfl_xor(rs, m, 32);
      lrow[r] = lrow[r] * alpha + rs;
      mrow[r] = mnew;
      o0[r] *= alpha; o1[r] *= alpha; o2[r] *= alpha; o3[r] *= alpha;
      _Float16* pp = &ps[(wave * 16 + (hid ? 8 + r : r)) * 64 + l16];
      pp[0] = (_Float16)p0; pp[16] = (_Float16)p1;
      pp[32] = (_Float16)p2; pp[48] = (_Float16)p3;
    }

    // O += P @ V  (K-contraction over 64 keys; per-wave ps, wave-order DS safe)
    #pragma unroll
    for (int kc = 0; kc < 64; kc += 32) {
      v16h ap = load_frag_a(ps, wave * 16 + l16, 64, kc, hid);
      v16h vb0 = load_frag_b(vt, 0 + l16, 64, kc, hid);
      v16h vb1 = load_frag_b(vt, 16 + l16, 64, kc, hid);
      v16h vb2 = load_frag_b(vt, 32 + l16, 64, kc, hid);
      v16h vb3 = load_frag_b(vt, 48 + l16, 64, kc, hid);
      o0 = wmma16(ap, vb0, o0);
      o1 = wmma16(ap, vb1, o1);
      o2 = wmma16(ap, vb2, o2);
      o3 = wmma16(ap, vb3, o3);
    }
    __syncthreads();
  }

  #pragma unroll
  for (int r = 0; r < 8; ++r) {
    const float inv = 1.0f / lrow[r];
    const int row = q0 + wave * 16 + (hid ? 8 + r : r);
    const size_t ob = base + (size_t)row * rstr;
    O[ob + 0 + l16]  = (_Float16)(o0[r] * inv);
    O[ob + 16 + l16] = (_Float16)(o1[r] * inv);
    O[ob + 32 + l16] = (_Float16)(o2[r] * inv);
    O[ob + 48 + l16] = (_Float16)(o3[r] * inv);
  }
}

// ------------------------------- launcher -----------------------------------
extern "C" void kernel_launch(void* const* d_in, const int* in_sizes, int n_in,
                              void* d_out, int out_size, void* d_ws, size_t ws_size,
                              hipStream_t stream) {
  const int Bb = 8, S = 1024, E = 1024, H = 16, D = 64, F = 4096;
  const int M = Bb * S;        // 8192 token rows
  const int HD = H * D;        // 1024

  const float* latents   = (const float*)d_in[0];
  const float* ln1_scale = (const float*)d_in[1];
  const float* wq        = (const float*)d_in[2];
  const float* wk        = (const float*)d_in[3];
  const float* wv        = (const float*)d_in[4];
  const float* q_ln      = (const float*)d_in[5];
  const float* k_ln      = (const float*)d_in[6];
  const float* wo        = (const float*)d_in[7];
  const float* ln2_scale = (const float*)d_in[8];
  const float* wi        = (const float*)d_in[9];
  const float* wo_mlp    = (const float*)d_in[10];
  float* out = (float*)d_out;

  char* ws = (char*)d_ws;
  const size_t MB = (size_t)1 << 20;
  _Float16* wq_h  = (_Float16*)(ws + 0 * MB);     // E*HD      (2 MB)
  _Float16* wk_h  = (_Float16*)(ws + 2 * MB);
  _Float16* wv_h  = (_Float16*)(ws + 4 * MB);
  _Float16* wo_h  = (_Float16*)(ws + 6 * MB);     // HD*E
  _Float16* wi_h  = (_Float16*)(ws + 8 * MB);     // E*F       (8 MB)
  _Float16* wo2_h = (_Float16*)(ws + 16 * MB);    // F*E       (8 MB)
  _Float16* xh    = (_Float16*)(ws + 24 * MB);    // M*E f16   (16 MB)
  _Float16* qh    = (_Float16*)(ws + 40 * MB);    // M*HD f16
  _Float16* kh    = (_Float16*)(ws + 56 * MB);
  _Float16* vh    = (_Float16*)(ws + 72 * MB);
  _Float16* oh    = (_Float16*)(ws + 88 * MB);
  float*    xres  = (float*)(ws + 104 * MB);      // M*E f32   (32 MB)
  _Float16* hh    = (_Float16*)(ws + 136 * MB);   // M*F f16   (64 MB)
  (void)in_sizes; (void)n_in; (void)out_size; (void)ws_size;

  // weights -> f16
  f2h_kernel<<<(E * HD + 255) / 256, 256, 0, stream>>>(wq, wq_h, E * HD);
  f2h_kernel<<<(E * HD + 255) / 256, 256, 0, stream>>>(wk, wk_h, E * HD);
  f2h_kernel<<<(E * HD + 255) / 256, 256, 0, stream>>>(wv, wv_h, E * HD);
  f2h_kernel<<<(HD * E + 255) / 256, 256, 0, stream>>>(wo, wo_h, HD * E);
  f2h_kernel<<<(E * F + 255) / 256, 256, 0, stream>>>(wi, wi_h, E * F);
  f2h_kernel<<<(F * E + 255) / 256, 256, 0, stream>>>(wo_mlp, wo2_h, F * E);

  // x = rmsnorm(latents) * ln1_scale  (f16)
  rmsnorm_rows<<<M, 256, 0, stream>>>(latents, ln1_scale, xh, E);

  // q,k,v = x @ w{q,k,v}   (128x128 tiles)
  gemm_kernel<0><<<dim3(HD / 128, M / 128), 256, 0, stream>>>(xh, wq_h, nullptr, nullptr, qh, M, HD, E);
  gemm_kernel<0><<<dim3(HD / 128, M / 128), 256, 0, stream>>>(xh, wk_h, nullptr, nullptr, kh, M, HD, E);
  gemm_kernel<0><<<dim3(HD / 128, M / 128), 256, 0, stream>>>(xh, wv_h, nullptr, nullptr, vh, M, HD, E);

  // per-head qk norm (q also gets 1/sqrt(D))
  qk_norm_kernel<<<M, 256, 0, stream>>>(qh, q_ln, 0.125f);
  qk_norm_kernel<<<M, 256, 0, stream>>>(kh, k_ln, 1.0f);

  // attention
  attn_kernel<<<dim3(S / 64, H, Bb), 128, 0, stream>>>(qh, kh, vh, oh, S, H);

  // x = o @ wo + latents  (f32 residual)
  gemm_kernel<1><<<dim3(E / 128, M / 128), 256, 0, stream>>>(oh, wo_h, latents, xres, nullptr, M, E, HD);

  // MLP: y = rmsnorm(x); h = gelu(y @ wi); out = h @ wo_mlp + x
  rmsnorm_rows<<<M, 256, 0, stream>>>(xres, ln2_scale, xh, E);
  gemm_kernel<2><<<dim3(F / 128, M / 128), 256, 0, stream>>>(xh, wi_h, nullptr, nullptr, hh, M, F, E);
  gemm_kernel<1><<<dim3(E / 128, M / 128), 256, 0, stream>>>(hh, wo2_h, xres, out, nullptr, M, E, F);
}
